// LinearAttention_1623497638245
// MI455X (gfx1250) — matrix operand order repaired and checked by reading, not compiled, not run on hardware
//
#include <hip/hip_runtime.h>
#include <hip/hip_bf16.h>
#include <stdint.h>

// ---------- types ----------
typedef __bf16 bf16x16 __attribute__((ext_vector_type(16)));
typedef float  f32x8   __attribute__((ext_vector_type(8)));
typedef unsigned int v4u __attribute__((ext_vector_type(4)));
typedef int          v8i __attribute__((ext_vector_type(8)));
typedef int          v4i __attribute__((ext_vector_type(4)));

union Frag16 {           // 32 bytes: one WMMA bf16 A/B operand
    bf16x16 v;
    uint4   q[2];
};

__device__ __forceinline__ unsigned short f32_to_bf16(float f) {
    union { float f; unsigned u; } x; x.f = f;
    unsigned r = x.u + 0x7FFFu + ((x.u >> 16) & 1u);   // round-to-nearest-even
    return (unsigned short)(r >> 16);
}

__device__ __forceinline__ f32x8 wmma_bf16(const bf16x16& a, const bf16x16& b, const f32x8& c) {
    return __builtin_amdgcn_wmma_f32_16x16x32_bf16(false, a, false, b, (short)0, c, false, false);
}

__device__ __forceinline__ void load_frag(Frag16& f, const unsigned short* p0, const unsigned short* p1) {
    f.q[0] = *(const uint4*)p0;
    f.q[1] = *(const uint4*)p1;
}

// LDS byte offset of a shared-memory pointer (flat addr low 32 bits = LDS offset)
__device__ __forceinline__ unsigned lds_off(const void* p) {
    return (unsigned)(uintptr_t)p;
}

// ---------- TDM: 2-D tile DMA global -> LDS (descriptor per CDNA5 ISA) ----------
// 6-arg builtin form: (uint32x4 g0, int32x8 g1, int32x4 g2, int32x4 g3, int32x8 g4, i32 cpol)
__device__ __forceinline__ void tdm_load_2d(unsigned lds_byte_off, const void* gptr,
                                            unsigned tensor_d0, unsigned tensor_d1,
                                            unsigned stride0,
                                            unsigned tile_d0, unsigned tile_d1) {
    unsigned long long ga = (unsigned long long)(uintptr_t)gptr;
    // group0: [1:0]=count=1, [63:32]=lds_addr, [120:64]=global_addr, [127:126]=type=2
    v4u g0 = { 1u,
               lds_byte_off,
               (unsigned)ga,
               (unsigned)((ga >> 32) & 0x01FFFFFFull) | 0x80000000u };
    // group1: [17:16]=data_size(1 -> 2B), [79:48]=tensor_dim0, [111:80]=tensor_dim1,
    //         [127:112]=tile_dim0, [143:128]=tile_dim1, [207:160]=tensor_dim0_stride
    v8i g1 = { (int)(1u << 16),
               (int)(tensor_d0 << 16),
               (int)((tensor_d0 >> 16) | (tensor_d1 << 16)),
               (int)((tensor_d1 >> 16) | (tile_d0 << 16)),
               (int)tile_d1,
               (int)stride0,
               0, 0 };
    v4i z4 = { 0, 0, 0, 0 };
    v8i z8 = { 0, 0, 0, 0, 0, 0, 0, 0 };
    __builtin_amdgcn_tensor_load_to_lds(g0, g1, z4, z4, z8, 0);
}

// ---------- constants (match reference) ----------
#define BB 8
#define NN 2048
#define MM 2048
#define DD 256

// ---------- fp32 -> bf16 convert ----------
__global__ void cvt_kernel(const float* __restrict__ src, unsigned short* __restrict__ dst, int n) {
    int i = blockIdx.x * blockDim.x + threadIdx.x;
    int stride = gridDim.x * blockDim.x;
    for (; i < n; i += stride) dst[i] = f32_to_bf16(src[i]);
}

// ---------- Fi [b][m][d] f32 -> FiT [b][d][m] bf16 (LDS-tiled transpose) ----------
__global__ __launch_bounds__(256) void transpose_kernel(const float* __restrict__ Fi,
                                                        unsigned short* __restrict__ ftb) {
    __shared__ float tile[32][33];
    int b  = blockIdx.z;
    int m0 = blockIdx.y * 32;
    int d0 = blockIdx.x * 32;
    int tx = threadIdx.x, ty = threadIdx.y;      // 32 x 8
    const float* src = Fi + ((size_t)b * MM + m0) * DD + d0;
#pragma unroll
    for (int i = 0; i < 32; i += 8) tile[ty + i][tx] = src[(size_t)(ty + i) * DD + tx];
    __syncthreads();
    unsigned short* dst = ftb + ((size_t)b * DD + d0) * MM + m0;
#pragma unroll
    for (int i = 0; i < 32; i += 8) dst[(size_t)(ty + i) * MM + tx] = f32_to_bf16(tile[tx][ty + i]);
}

// ---------- projection: q = x @ W^T + b  (WMMA, software-pipelined B frags) ----------
// block = 128 (4 waves); block handles 16 rows, wave w handles 64 output cols (4 tiles).
__global__ __launch_bounds__(128) void proj_kernel(const unsigned short* __restrict__ xb,
                                                   const unsigned short* __restrict__ wb,
                                                   const float* __restrict__ bias,
                                                   unsigned short* __restrict__ qb) {
    int lane = threadIdx.x & 31;
    int wave = threadIdx.x >> 5;
    int half = lane >> 4, l15 = lane & 15;
    int row0 = blockIdx.x * 16;

    // A fragments: rows row0..row0+15, all of K=256 (8 frags)
    Frag16 a[8];
    const unsigned short* xrow = xb + (size_t)(row0 + l15) * DD;
#pragma unroll
    for (int kk = 0; kk < 8; ++kk)
        load_frag(a[kk], xrow + kk * 32 + half * 16, xrow + kk * 32 + half * 16 + 8);

    int e0 = wave * 64;
    const unsigned short* wbase = wb + half * 16;   // + e*DD + kk*32 below
    f32x8 acc[4];
#pragma unroll
    for (int t = 0; t < 4; ++t) {
        float bv = bias[e0 + t * 16 + l15];
#pragma unroll
        for (int r = 0; r < 8; ++r) acc[t][r] = bv;
    }

    // ping-pong double buffer of the 4 W fragments per k-step:
    // loads for step kk+1 are issued before the 4 WMMAs of step kk consume buffer kk&1
    Frag16 bb[2][4];
#pragma unroll
    for (int t = 0; t < 4; ++t) {
        const unsigned short* p = wbase + (size_t)(e0 + t * 16 + l15) * DD;
        load_frag(bb[0][t], p, p + 8);
    }
#pragma unroll
    for (int kk = 0; kk < 8; ++kk) {
        int cu = kk & 1, nx = cu ^ 1;
        if (kk < 7) {
#pragma unroll
            for (int t = 0; t < 4; ++t) {
                const unsigned short* p = wbase + (size_t)(e0 + t * 16 + l15) * DD + (kk + 1) * 32;
                load_frag(bb[nx][t], p, p + 8);
            }
        }
#pragma unroll
        for (int t = 0; t < 4; ++t)
            acc[t] = wmma_bf16(a[kk].v, bb[cu][t].v, acc[t]);
    }

#pragma unroll
    for (int t = 0; t < 4; ++t) {
        int e = e0 + t * 16 + l15;
#pragma unroll
        for (int r = 0; r < 8; ++r)
            qb[(size_t)(row0 + r + half * 8) * DD + e] = f32_to_bf16(acc[t][r]);
    }
}

// ---------- fused flash-attention with TDM double-buffered K/V staging ----------
// block = 128 (4 waves); each wave owns 16 query rows x full D=256 output.
__global__ __launch_bounds__(128) void attn_kernel(const unsigned short* __restrict__ qb,
                                                   const unsigned short* __restrict__ fb,
                                                   const unsigned short* __restrict__ ftb,
                                                   float* __restrict__ out) {
    __shared__ unsigned short q_lds[4][16 * DD];     // 32 KB : per-wave q tile
    __shared__ unsigned short kbuf[2][32 * DD];      // 32 KB : K chunk  [key][d], dbl-buffered
    __shared__ unsigned short vbuf[2][DD * 32];      // 32 KB : V^T chunk [d][key], dbl-buffered
    __shared__ unsigned short p_lds[4][16][32];      //  4 KB : C->A layout bounce for P

    int lane = threadIdx.x & 31;
    int wave = threadIdx.x >> 5;
    int half = lane >> 4, l15 = lane & 15;
    int tile = blockIdx.x * 4 + wave;
    int row0 = tile * 16;                            // row among B*N
    int b    = row0 / NN;

    const unsigned short* fbB = fb  + (size_t)b * MM * DD;   // K  : [m][d] bf16
    const unsigned short* ftB = ftb + (size_t)b * DD * MM;   // V^T: [d][m] bf16

    // stage q tile (16 rows x 256, contiguous 8KB) into LDS
    {
        const uint4* src = (const uint4*)(qb + (size_t)row0 * DD);
        uint4* dst = (uint4*)q_lds[wave];
#pragma unroll
        for (int t = 0; t < 16; ++t) dst[t * 32 + lane] = src[t * 32 + lane];
    }

    // TDM prologue: chunk 0 -> buffer 0
    if (wave == 0) {
        tdm_load_2d(lds_off(&kbuf[0][0]), fbB, DD, MM, DD, DD, 32);   // 32 rows x 256
        tdm_load_2d(lds_off(&vbuf[0][0]), ftB, MM, DD, MM, 32, DD);   // 256 rows x 32
    }

    f32x8 o[16];
#pragma unroll
    for (int t = 0; t < 16; ++t)
#pragma unroll
        for (int r = 0; r < 8; ++r) o[t][r] = 0.0f;

    float m_r[8], l_r[8];
#pragma unroll
    for (int r = 0; r < 8; ++r) { m_r[r] = -1.0e30f; l_r[r] = 0.0f; }

    const unsigned short* qrow = q_lds[wave] + (size_t)l15 * DD;

    for (int c = 0; c < MM / 32; ++c) {
        int j   = c * 32;
        int cur = c & 1;
        if (wave == 0) {
            if (c + 1 < MM / 32) {     // prefetch next chunk into other buffer, then
                int nxt = cur ^ 1;     // wait for the 2 outstanding loads of chunk c
                tdm_load_2d(lds_off(&kbuf[nxt][0]), fbB + (size_t)(j + 32) * DD, DD, MM, DD, DD, 32);
                tdm_load_2d(lds_off(&vbuf[nxt][0]), ftB + (size_t)(j + 32),      MM, DD, MM, 32, DD);
                __builtin_amdgcn_s_wait_tensorcnt(2);
            } else {
                __builtin_amdgcn_s_wait_tensorcnt(0);
            }
        }
        __syncthreads();               // chunk c data visible to all waves

        const unsigned short* kb = kbuf[cur];   // [32][256]
        const unsigned short* vb = vbuf[cur];   // [256][32]

        // ---- scores S[16 x 32] = q_tile @ K_chunk^T, ping-pong pipelined from LDS ----
        f32x8 s0, s1;
#pragma unroll
        for (int r = 0; r < 8; ++r) { s0[r] = 0.0f; s1[r] = 0.0f; }

        Frag16 af[2], b0f[2], b1f[2];
        load_frag(af[0],  qrow + half * 16,                      qrow + half * 16 + 8);
        load_frag(b0f[0], kb + (size_t)l15 * DD + half * 16,     kb + (size_t)l15 * DD + half * 16 + 8);
        load_frag(b1f[0], kb + (size_t)(16 + l15) * DD + half * 16,
                          kb + (size_t)(16 + l15) * DD + half * 16 + 8);
#pragma unroll
        for (int kk = 0; kk < 8; ++kk) {
            int cu = kk & 1, nx = cu ^ 1;
            if (kk < 7) {
                int k1 = (kk + 1) * 32;
                load_frag(af[nx],  qrow + k1 + half * 16, qrow + k1 + half * 16 + 8);
                const unsigned short* kr0 = kb + (size_t)l15 * DD + k1 + half * 16;
                const unsigned short* kr1 = kb + (size_t)(16 + l15) * DD + k1 + half * 16;
                load_frag(b0f[nx], kr0, kr0 + 8);
                load_frag(b1f[nx], kr1, kr1 + 8);
            }
            s0 = wmma_bf16(af[cu].v, b0f[cu].v, s0);
            s1 = wmma_bf16(af[cu].v, b1f[cu].v, s1);
        }

        // ---- online softmax; row r lives across 16 lanes of one VGPR half ----
        float alpha[8];
#pragma unroll
        for (int r = 0; r < 8; ++r) {
            float mx = fmaxf(s0[r], s1[r]);
#pragma unroll
            for (int msk = 8; msk >= 1; msk >>= 1) mx = fmaxf(mx, __shfl_xor(mx, msk, 32));
            float mn = fmaxf(m_r[r], mx);
            alpha[r] = __expf(m_r[r] - mn);
            m_r[r] = mn;
            float p0 = __expf(s0[r] - mn);
            float p1 = __expf(s1[r] - mn);
            float sum = p0 + p1;
#pragma unroll
            for (int msk = 8; msk >= 1; msk >>= 1) sum += __shfl_xor(sum, msk, 32);
            l_r[r] = l_r[r] * alpha[r] + sum;
            p_lds[wave][r + half * 8][l15]      = f32_to_bf16(p0);
            p_lds[wave][r + half * 8][l15 + 16] = f32_to_bf16(p1);
        }

        // reload P as a 16x32 bf16 A fragment (same-wave DS ops are in order)
        Frag16 pa;
        load_frag(pa, &p_lds[wave][l15][half * 16], &p_lds[wave][l15][half * 16 + 8]);

        // ---- O = O*alpha + P @ V_chunk, ping-pong pipelined V frags from LDS ----
        Frag16 vf[2];
        load_frag(vf[0], vb + (size_t)l15 * 32 + half * 16,
                         vb + (size_t)l15 * 32 + half * 16 + 8);
#pragma unroll
        for (int t = 0; t < 16; ++t) {
            int cu = t & 1, nx = cu ^ 1;
            if (t < 15) {
                const unsigned short* vr = vb + (size_t)((t + 1) * 16 + l15) * 32 + half * 16;
                load_frag(vf[nx], vr, vr + 8);
            }
#pragma unroll
            for (int r = 0; r < 8; ++r) o[t][r] *= alpha[r];
            o[t] = wmma_bf16(pa.v, vf[cu].v, o[t]);
        }

        __syncthreads();               // everyone done with buffer before TDM overwrites it
    }

    // ---- finalize: divide by row sums, store fp32 ----
#pragma unroll
    for (int t = 0; t < 16; ++t) {
#pragma unroll
        for (int r = 0; r < 8; ++r) {
            float v = o[t][r] / l_r[r];
            out[(size_t)(row0 + r + half * 8) * DD + t * 16 + l15] = v;
        }
    }
}

// ---------- host launcher ----------
extern "C" void kernel_launch(void* const* d_in, const int* in_sizes, int n_in,
                              void* d_out, int out_size, void* d_ws, size_t ws_size,
                              hipStream_t stream) {
    const float* x    = (const float*)d_in[0];   // [B,N,D]
    const float* Fi   = (const float*)d_in[1];   // [B,M,D]
    const float* W    = (const float*)d_in[2];   // [D,D]
    const float* bias = (const float*)d_in[3];   // [D]
    float* out = (float*)d_out;

    const size_t nx = (size_t)BB * NN * DD;
    const size_t nf = (size_t)BB * MM * DD;
    const size_t nw = (size_t)DD * DD;

    unsigned short* xb  = (unsigned short*)d_ws;  // x  bf16 [B*N][D]
    unsigned short* qb  = xb  + nx;               // q  bf16 [B*N][D]
    unsigned short* fbb = qb  + nx;               // Fi bf16 [B][M][D]
    unsigned short* ftb = fbb + nf;               // Fi bf16 transposed [B][D][M]
    unsigned short* wb  = ftb + nf;               // W  bf16 [D][D]

    cvt_kernel<<<2048, 256, 0, stream>>>(x,  xb,  (int)nx);
    cvt_kernel<<<2048, 256, 0, stream>>>(Fi, fbb, (int)nf);
    cvt_kernel<<<256,  256, 0, stream>>>(W,  wb,  (int)nw);
    transpose_kernel<<<dim3(DD / 32, MM / 32, BB), dim3(32, 8), 0, stream>>>(Fi, ftb);

    proj_kernel<<<(BB * NN) / 16, 128, 0, stream>>>(xb, wb, bias, qb);
    attn_kernel<<<(BB * NN) / 64, 128, 0, stream>>>(qb, fbb, ftb, out);
}
